// MoEGroupedLinear_10402410791790
// MI455X (gfx1250) — matrix operand order repaired and checked by reading, not compiled, not run on hardware
//
#include <hip/hip_runtime.h>
#include <math.h>

// MI455X / gfx1250: wave32, WMMA bf16 16x16x32, f32 accumulate.
// Double-buffered LDS, one barrier per k-step, register-staged prefetch.
typedef __attribute__((ext_vector_type(16))) __bf16 v16bf;
typedef __attribute__((ext_vector_type(2)))  __bf16 bf16x2;
typedef __attribute__((ext_vector_type(8)))  float  v8f;

#define BM 128
#define BN 128
#define BK 32
#define LDS_STRIDE 40   // bf16 elems per row = 80 bytes: 16B aligned, bank-skewed

union FragBF {
  v16bf v;
  uint4 q[2];
};

static __device__ __forceinline__ unsigned pk_bf16(float a, float b) {
  bf16x2 v;
  v.x = (__bf16)a;
  v.y = (__bf16)b;
  return __builtin_bit_cast(unsigned, v);   // -> v_cvt_pk_bf16_f32
}

static __device__ __forceinline__ uint4 cvt4(const float4 f0, const float4 f1) {
  uint4 q;
  q.x = pk_bf16(f0.x, f0.y);
  q.y = pk_bf16(f0.z, f0.w);
  q.z = pk_bf16(f1.x, f1.y);
  q.w = pk_bf16(f1.z, f1.w);
  return q;
}

__global__ __launch_bounds__(256)
void moe_grouped_gemm_gelu(const float* __restrict__ x,
                           const int*   __restrict__ cnt,
                           const float* __restrict__ w,
                           const float* __restrict__ bias,
                           float* __restrict__ out,
                           int T, int D_IN, int D_OUT, int E)
{
  __shared__ __align__(16) __bf16 As[2][BM * LDS_STRIDE];
  __shared__ __align__(16) __bf16 Bs[2][BN * LDS_STRIDE];

  const int e = blockIdx.z;
  int seg_start = 0;
  for (int i = 0; i < e; ++i) seg_start += cnt[i];
  const int cnt_e = cnt[e];

  const int m_tile = blockIdx.y * BM;     // row offset inside this expert's segment
  if (m_tile >= cnt_e) return;            // uniform exit: EXEC stays all-1s for WMMA

  const int n0     = blockIdx.x * BN;
  const int m_base = seg_start + m_tile;  // global token row of tile origin

  const int tid  = threadIdx.x;
  const int lane = tid & 31;
  const int wave = tid >> 5;
  const int wm   = wave & 3;              // 4 waves along M (32 rows each)
  const int wn   = wave >> 2;             // 2 waves along N (64 cols each)
  const int lm   = lane & 15;
  const int lhi  = lane >> 4;             // lane half select

  // --- staging roles: 256 threads cover 128 rows x 2 half-rows (16 f32 each) ---
  const int srow  = tid >> 1;             // 0..127
  const int shalf = tid & 1;              // 0 or 1 (16-col half)

  // Clamp source rows into valid memory instead of zero-filling:
  //  * a garbage A-row only affects the same (never-stored) output row
  //  * a garbage B-row only affects never-stored output columns
  const int arow = min(m_base + srow, seg_start + cnt_e - 1);
  const int brow = min(n0 + srow, D_OUT - 1);
  const size_t w_base = (size_t)e * D_OUT * D_IN;
  const float* srcA = x + (size_t)arow * D_IN + shalf * 16;
  const float* srcB = w + w_base + (size_t)brow * D_IN + shalf * 16;

  const int sOff = srow * LDS_STRIDE + shalf * 16;   // staging store offset (elems)
  const int fAOff = (wm * 32 + lm) * LDS_STRIDE;     // frag base rows (elems)
  const int fBOff = (wn * 64 + lm) * LDS_STRIDE;

  v8f acc[2][4] = {};
  float4 ra[4], rb[4];

  // ---- prologue: tile 0 -> regs -> LDS buf0; prefetch tile 1 ----
  #pragma unroll
  for (int j = 0; j < 4; ++j) {
    ra[j] = ((const float4*)srcA)[j];
    rb[j] = ((const float4*)srcB)[j];
  }
  {
    uint4* dA = (uint4*)&As[0][sOff];
    uint4* dB = (uint4*)&Bs[0][sOff];
    dA[0] = cvt4(ra[0], ra[1]);
    dA[1] = cvt4(ra[2], ra[3]);
    dB[0] = cvt4(rb[0], rb[1]);
    dB[1] = cvt4(rb[2], rb[3]);
  }
  const int kLast = D_IN - BK;
  {
    const int k1 = min(BK, kLast);
    const float* pa = srcA + k1;
    const float* pb = srcB + k1;
    #pragma unroll
    for (int j = 0; j < 4; ++j) {
      ra[j] = ((const float4*)pa)[j];
      rb[j] = ((const float4*)pb)[j];
    }
  }
  __syncthreads();

  // One pipeline step: store tile (it+1) regs->buf[nxt], prefetch tile (it+2),
  // compute WMMA from buf[cur], barrier. `cur` is a literal at each call site
  // so all LDS addresses fold to immediate offsets.
  auto step = [&](int kPrefetch, int cur) {
    const int nxt = cur ^ 1;
    // store next tile (converts consume prefetched regs; loadcnt wait lands here)
    {
      uint4* dA = (uint4*)&As[nxt][sOff];
      uint4* dB = (uint4*)&Bs[nxt][sOff];
      dA[0] = cvt4(ra[0], ra[1]);
      dA[1] = cvt4(ra[2], ra[3]);
      dB[0] = cvt4(rb[0], rb[1]);
      dB[1] = cvt4(rb[2], rb[3]);
    }
    // prefetch tile it+2 (clamped; dead-but-safe near the end)
    {
      const float* pa = srcA + kPrefetch;
      const float* pb = srcB + kPrefetch;
      #pragma unroll
      for (int j = 0; j < 4; ++j) {
        ra[j] = ((const float4*)pa)[j];
        rb[j] = ((const float4*)pb)[j];
      }
    }
    // ---- A fragments: 16x32 bf16; per-lane K {0-7,16-23} / {8-15,24-31} ----
    FragBF a[2];
    #pragma unroll
    for (int mi = 0; mi < 2; ++mi) {
      const char* row = (const char*)&As[cur][fAOff + mi * 16 * LDS_STRIDE] + lhi * 16;
      a[mi].q[0] = *(const uint4*)(row);
      a[mi].q[1] = *(const uint4*)(row + 32);
    }
    // ---- B fragments: 32x16 bf16; per-lane K {0-7,16-23} / {8-15,24-31} ----
    FragBF b[4];
    #pragma unroll
    for (int ni = 0; ni < 4; ++ni) {
      const char* row = (const char*)&Bs[cur][fBOff + ni * 16 * LDS_STRIDE] + lhi * 16;
      b[ni].q[0] = *(const uint4*)(row);
      b[ni].q[1] = *(const uint4*)(row + 32);
    }
    #pragma unroll
    for (int mi = 0; mi < 2; ++mi)
      #pragma unroll
      for (int ni = 0; ni < 4; ++ni)
        acc[mi][ni] = __builtin_amdgcn_wmma_f32_16x16x32_bf16(
            false, a[mi].v, false, b[ni].v, (short)0, acc[mi][ni], false, false);
    __syncthreads();
  };

  const int NIT = D_IN / BK;
  for (int it = 0; it + 1 < NIT; it += 2) {
    step(min((it + 2) * BK, kLast), 0);
    step(min((it + 3) * BK, kLast), 1);
  }
  if (NIT & 1) step(kLast, 0);

  // ---- epilogue: bias + exact GELU, f32 store ----
  // C/D layout: VGPR r, lanes 0-15 -> M=r, N=lane; lanes 16-31 -> M=r+8, N=lane-16
  float bias_v[4];
  int   ncol[4];
  #pragma unroll
  for (int ni = 0; ni < 4; ++ni) {
    ncol[ni]   = n0 + wn * 64 + ni * 16 + lm;
    bias_v[ni] = bias[(size_t)e * D_OUT + min(ncol[ni], D_OUT - 1)];
  }

  #pragma unroll
  for (int mi = 0; mi < 2; ++mi) {
    #pragma unroll
    for (int r = 0; r < 8; ++r) {
      const int m_local = wm * 32 + mi * 16 + r + (lhi << 3);
      const bool mvalid = (m_tile + m_local) < cnt_e;
      const size_t orow = (size_t)(m_base + m_local) * D_OUT;
      #pragma unroll
      for (int ni = 0; ni < 4; ++ni) {
        if (mvalid && ncol[ni] < D_OUT) {
          float v = acc[mi][ni][r] + bias_v[ni];
          v = 0.5f * v * (1.0f + erff(v * 0.70710678118654752f));
          out[orow + ncol[ni]] = v;
        }
      }
    }
  }
}

extern "C" void kernel_launch(void* const* d_in, const int* in_sizes, int n_in,
                              void* d_out, int out_size, void* d_ws, size_t ws_size,
                              hipStream_t stream) {
  const float* x    = (const float*)d_in[0];
  const int*   cnt  = (const int*)  d_in[1];
  const float* w    = (const float*)d_in[2];
  const float* bias = (const float*)d_in[3];
  float* out = (float*)d_out;

  const int E     = in_sizes[1];
  const int D_OUT = in_sizes[3] / E;
  const long long wsz = (long long)in_sizes[2];
  const int D_IN  = (int)(wsz / ((long long)E * D_OUT));
  const int T     = in_sizes[0] / D_IN;

  dim3 grid((D_OUT + BN - 1) / BN, (T + BM - 1) / BM, E);
  moe_grouped_gemm_gelu<<<grid, 256, 0, stream>>>(x, cnt, w, bias, out,
                                                  T, D_IN, D_OUT, E);
}
